// STAR_63127429316724
// MI455X (gfx1250) — matrix operand order repaired and checked by reading, not compiled, not run on hardware
//
#include <hip/hip_runtime.h>
#include <hip/hip_bf16.h>

typedef __attribute__((ext_vector_type(16))) _Float16 v16h;
typedef __attribute__((ext_vector_type(8)))  _Float16 v8h;
typedef __attribute__((ext_vector_type(8)))  float    v8f;

#define B_   256
#define S_   50
#define H_   100
#define KPAD 128
#define HH_  10000
#define ITEMS_ 50000
#define NPAD_IC   10048   // 157*64
#define NPAD_PRED 50048   // 782*64

__device__ __forceinline__ float sig(float x) { return 1.0f / (1.0f + __expf(-x)); }

__device__ __forceinline__ v16h cat8(v8h lo, v8h hi) {
  return __builtin_shufflevector(lo, hi, 0, 1, 2, 3, 4, 5, 6, 7,
                                         8, 9, 10, 11, 12, 13, 14, 15);
}

// ---------------- h3 precompute: H3f16[(t*B+b)*128 + k] ----------------
__global__ __launch_bounds__(256) void h3_kernel(const float* __restrict__ cx,
                                                 const float* __restrict__ wix,
                                                 const float* __restrict__ bix,
                                                 const float* __restrict__ bh3,
                                                 _Float16* __restrict__ H3) {
  int idx = blockIdx.x * blockDim.x + threadIdx.x;   // over S*B*128
  if (idx >= S_ * B_ * KPAD) return;
  int k = idx & (KPAD - 1);
  int bt = idx >> 7;          // t*B + b
  int t = bt / B_, b = bt - t * B_;
  float v = 0.0f;
  if (k < H_) {
    float c = cx[b * S_ + t];
    v = sig(c * wix[k] + bix[k] + bh3[k]);
  }
  H3[idx] = (_Float16)v;
}

// ---- weight pad+convert: W[n][k(100)] f32 -> Wh[n][128] f16 (zero-padded) ----
__global__ __launch_bounds__(256) void wpad_kernel(const float* __restrict__ W,
                                                   _Float16* __restrict__ Wh,
                                                   int Nlog, int NPAD) {
  int idx = blockIdx.x * blockDim.x + threadIdx.x;   // over NPAD*128
  if (idx >= NPAD * KPAD) return;
  int n = idx >> 7;
  int k = idx & (KPAD - 1);
  float v = 0.0f;
  if (k < H_ && n < Nlog) v = W[(size_t)n * H_ + k];
  Wh[idx] = (_Float16)v;
}

// ---------------- part1[t,b,j] = emb[item_x[b,t]] @ Wx^T + bx ----------------
__global__ __launch_bounds__(256) void part1_kernel(const int* __restrict__ item_x,
                                                    const float* __restrict__ emb,
                                                    const float* __restrict__ Wx,
                                                    const float* __restrict__ bx,
                                                    float* __restrict__ P1) {
  int idx = blockIdx.x * blockDim.x + threadIdx.x;   // over S*B*100
  if (idx >= S_ * B_ * H_) return;
  int j = idx % H_;
  int bt = idx / H_;
  int t = bt / B_, b = bt - t * B_;
  int item = item_x[b * S_ + t];
  const float* e = emb + (size_t)item * H_;
  const float* w = Wx + (size_t)j * H_;
  float acc = bx[j];
#pragma unroll 4
  for (int k = 0; k < H_; ++k) acc += e[k] * w[k];
  P1[idx] = acc;
}

__global__ __launch_bounds__(256) void zero_h(float* __restrict__ h) {
  int idx = blockIdx.x * blockDim.x + threadIdx.x;
  if (idx < B_ * H_) h[idx] = 0.0f;
}

// ---------------- WMMA GEMM, K=128 (f16), bias + optional sigmoid ----------------
// out[m, n] = act( sum_k A[m][k] * Bh[n][k] + bias[n] ),  M fixed = 256.
// Block: 256 thr = 8 waves stacked on M (covers 128 rows) x 64 cols.
// B panel (64x128 f16 = 16 KB) staged into LDS with CDNA5 async DMA
// (global_load_async_to_lds_b128, ASYNCcnt) — no VGPR round-trip.
// Each wave: 4 A fragments (reused x4) x 4 B fragments -> 16 v_wmma.
__global__ __launch_bounds__(256) void gemm_k128(const _Float16* __restrict__ A,   // [256][128]
                                                 const _Float16* __restrict__ Bh,  // [NPAD][128]
                                                 const float* __restrict__ bias,   // [Nout]
                                                 float* __restrict__ out,          // [256][Nout]
                                                 int Nout, int applySig) {
  __shared__ _Float16 Bs[64 * KPAD];                 // 16 KB

  const int tid  = threadIdx.x;
  const int lane = tid & 31;
  const int wave = tid >> 5;                         // 0..7 -> M sub-tile
  const int g    = lane >> 4;                        // half-wave group
  const int ml   = lane & 15;

  const int m0 = blockIdx.y * 128 + wave * 16;
  const int n0 = blockIdx.x * 64;

  // --- async DMA: 64x128 f16 B panel -> LDS, 16 B per lane per op, 4 ops ---
  // INST_OFFSET advances both the LDS dest and the global src (ISA 8 §4.4),
  // and our LDS layout mirrors the global panel linearly, so one offset works.
  {
    unsigned ldsAddr = (unsigned)(size_t)(&Bs[0]) + (unsigned)tid * 16u;  // addr[31:0] = LDS offset
    unsigned long long gAddr =
        (unsigned long long)(const void*)(Bh + (size_t)n0 * KPAD) + (unsigned long long)tid * 16ull;
    asm volatile(
        "global_load_async_to_lds_b128 %0, %1, off\n\t"
        "global_load_async_to_lds_b128 %0, %1, off offset:4096\n\t"
        "global_load_async_to_lds_b128 %0, %1, off offset:8192\n\t"
        "global_load_async_to_lds_b128 %0, %1, off offset:12288\n\t"
        "s_wait_asynccnt 0"
        :: "v"(ldsAddr), "v"(gAddr) : "memory");
  }
  __syncthreads();

  // --- A fragments: lane ml = M row; elems 0..7 @ K=32kc+8g, 8..15 @ +16 ---
  const _Float16* arow = A + (size_t)(m0 + ml) * KPAD;
  v16h afrag[4];
#pragma unroll
  for (int kc = 0; kc < 4; ++kc) {
    int ka = kc * 32 + g * 8;
    afrag[kc] = cat8(*(const v8h*)(arow + ka), *(const v8h*)(arow + ka + 16));
  }

  // --- 4 C tiles per wave; B frag: lane ml = N col; elems 0..7 @ K=32kc+8g, 8..15 @ +16 ---
  v8f c[4] = {v8f{}, v8f{}, v8f{}, v8f{}};
#pragma unroll
  for (int nt = 0; nt < 4; ++nt) {
    const _Float16* brow = Bs + (size_t)(nt * 16 + ml) * KPAD;
#pragma unroll
    for (int kc = 0; kc < 4; ++kc) {
      int kb = kc * 32 + g * 8;
      v16h b = cat8(*(const v8h*)(brow + kb), *(const v8h*)(brow + kb + 16));
      c[nt] = __builtin_amdgcn_wmma_f32_16x16x32_f16(false, afrag[kc], false, b,
                                                     (short)0, c[nt], false, false);
    }
  }

  // --- epilogue: C layout M = r + 8*g, N = ml ---
#pragma unroll
  for (int nt = 0; nt < 4; ++nt) {
    int n = n0 + nt * 16 + ml;
    if (n < Nout) {
      float bia = bias[n];
#pragma unroll
      for (int r = 0; r < 8; ++r) {
        int m = m0 + r + 8 * g;
        float v = c[nt][r] + bia;
        if (applySig) v = sig(v);
        out[(size_t)m * Nout + n] = v;
      }
    }
  }
}

// ---------------- sequential recurrence update (one block per batch row) ----------------
// h[b,j] = sigmoid( part1[t,b,j] + sum_i h[b,i]*IC[b, i*100+j] + b_ic[j] )
__global__ __launch_bounds__(128) void contract_update(const float* __restrict__ IC,
                                                       const float* __restrict__ P1t,
                                                       const float* __restrict__ b_ic,
                                                       float* __restrict__ h) {
  __shared__ float hs[H_];
  int b = blockIdx.x, j = threadIdx.x;
  if (j < H_) hs[j] = h[b * H_ + j];
  __syncthreads();
  if (j < H_) {
    const float* icb = IC + (size_t)b * HH_ + j;
    float acc = 0.0f;
#pragma unroll 4
    for (int i = 0; i < H_; ++i) acc += hs[i] * icb[(size_t)i * H_];
    float v = P1t[b * H_ + j] + acc + b_ic[j];
    h[b * H_ + j] = sig(v);
  }
}

// ---------------- pad h (f32 [256,100]) into f16 [256,128] ----------------
__global__ __launch_bounds__(256) void hpad_kernel(const float* __restrict__ h,
                                                   _Float16* __restrict__ Hf) {
  int idx = blockIdx.x * blockDim.x + threadIdx.x;   // over 256*128
  if (idx >= B_ * KPAD) return;
  int b = idx >> 7, k = idx & 127;
  Hf[idx] = (k < H_) ? (_Float16)h[b * H_ + k] : (_Float16)0.0f;
}

extern "C" void kernel_launch(void* const* d_in, const int* in_sizes, int n_in,
                              void* d_out, int out_size, void* d_ws, size_t ws_size,
                              hipStream_t stream) {
  // Inputs per setup_inputs() order
  const int*   item_x = (const int*)  d_in[0];   // [B,S]
  const float* cx     = (const float*)d_in[1];   // [B,S]
  // d_in[2] user_id: unused by reference math
  const float* emb    = (const float*)d_in[3];   // [50000,100]
  // d_in[4] W_h3: multiplied by h2 == 0, dead
  const float* b_h3   = (const float*)d_in[5];
  const float* W_ix   = (const float*)d_in[6];   // [100,1]
  const float* b_ix   = (const float*)d_in[7];
  const float* W2     = (const float*)d_in[8];   // [10000,100]
  const float* b2     = (const float*)d_in[9];
  const float* b_ic   = (const float*)d_in[10];
  const float* Wx     = (const float*)d_in[11];  // [100,100]
  const float* bx     = (const float*)d_in[12];
  const float* W_pred = (const float*)d_in[13];  // [50000,100]
  const float* b_pred = (const float*)d_in[14];
  float* logits = (float*)d_out;                 // [256,50000]

  // Workspace carve-up (all chunk sizes multiples of 256 bytes)
  char* ws = (char*)d_ws;
  size_t off = 0;
  _Float16* H3   = (_Float16*)(ws + off); off += (size_t)S_ * B_ * KPAD * 2;      // 3,276,800
  _Float16* W2h  = (_Float16*)(ws + off); off += (size_t)NPAD_IC * KPAD * 2;      // 2,572,288
  _Float16* WPh  = (_Float16*)(ws + off); off += (size_t)NPAD_PRED * KPAD * 2;    // 12,812,288
  float*    P1   = (float*)   (ws + off); off += (size_t)S_ * B_ * H_ * 4;        // 5,120,000
  float*    IC   = (float*)   (ws + off); off += (size_t)B_ * HH_ * 4;            // 10,240,000
  float*    hbuf = (float*)   (ws + off); off += (size_t)B_ * H_ * 4;             // 102,400
  _Float16* Hf16 = (_Float16*)(ws + off); off += (size_t)B_ * KPAD * 2;           // 65,536

  // --- time-parallel precompute ---
  h3_kernel<<<(S_ * B_ * KPAD + 255) / 256, 256, 0, stream>>>(cx, W_ix, b_ix, b_h3, H3);
  wpad_kernel<<<(NPAD_IC * KPAD + 255) / 256, 256, 0, stream>>>(W2, W2h, HH_, NPAD_IC);
  wpad_kernel<<<(NPAD_PRED * KPAD + 255) / 256, 256, 0, stream>>>(W_pred, WPh, ITEMS_, NPAD_PRED);
  part1_kernel<<<(S_ * B_ * H_ + 255) / 256, 256, 0, stream>>>(item_x, emb, Wx, bx, P1);
  zero_h<<<(B_ * H_ + 255) / 256, 256, 0, stream>>>(hbuf);

  // --- sequential scan: heavy GEMM is time-independent, recurrence is tiny ---
  dim3 gIC(NPAD_IC / 64, B_ / 128);    // 157 x 2
  for (int t = 0; t < S_; ++t) {
    gemm_k128<<<gIC, 256, 0, stream>>>(H3 + (size_t)t * B_ * KPAD, W2h, b2, IC,
                                       HH_, 1);
    contract_update<<<B_, 128, 0, stream>>>(IC, P1 + (size_t)t * B_ * H_, b_ic, hbuf);
  }

  // --- prediction head ---
  hpad_kernel<<<(B_ * KPAD + 255) / 256, 256, 0, stream>>>(hbuf, Hf16);
  dim3 gP(NPAD_PRED / 64, B_ / 128);   // 782 x 2
  gemm_k128<<<gP, 256, 0, stream>>>(Hf16, WPh, b_pred, logits, ITEMS_, 0);
}
